// MotionEstimator_27771258536125
// MI455X (gfx1250) — matrix operand order repaired and checked by reading, not compiled, not run on hardware
//
#include <hip/hip_runtime.h>
#include <stdint.h>

typedef __attribute__((ext_vector_type(16))) _Float16 v16h;
typedef __attribute__((ext_vector_type(8)))  float    v8f;
typedef __attribute__((ext_vector_type(4)))  int      v4i_;

#define B_   8
#define H_   192
#define W_   192
#define C_   64     // feature channels
#define ND   81     // displacement channels (9x9)
#define NDP  96     // padded to multiple of 32 for WMMA K

// CDNA5 async global->LDS staging, guarded so we fall back to the proven
// synchronous path if this toolchain doesn't expose the builtins.
#if defined(__gfx1250__) && __has_builtin(__builtin_amdgcn_global_load_async_to_lds_b128) && __has_builtin(__builtin_amdgcn_s_wait_asynccnt)
#define ASYNC_STAGE 1
#else
#define ASYNC_STAGE 0
#endif

// ---------------------------------------------------------------------------
// Correlation: LiteFlowNet cost volume. 16(w) x 8(h) pixel tile per block,
// feat2 halo window staged in LDS as f16, feat1 pixel vector in packed regs,
// f32 accumulation. Output NHWC f16 with channels padded 81 -> 96 (zeros).
// ---------------------------------------------------------------------------
__global__ __launch_bounds__(128) void corr_kernel(const float* __restrict__ f1,
                                                   const float* __restrict__ f2,
                                                   _Float16* __restrict__ corr)
{
    const int HALO_H = 8 + 8;    // 16
    const int HALO_W = 16 + 8;   // 24
    __shared__ __align__(16) _Float16 s2[HALO_H * HALO_W * C_];   // 49152 B

    const int b  = blockIdx.z;
    const int y0 = blockIdx.y * 8;
    const int x0 = blockIdx.x * 16;
    const int tid = threadIdx.x;

    // feat2 halo: fp32 NCHW -> f16 pixel-major [py][px][c] in LDS
    for (int i = tid; i < HALO_H * HALO_W * C_; i += 128) {
        int px = i % HALO_W;
        int t  = i / HALO_W;
        int py = t % HALO_H;
        int c  = t / HALO_H;
        int y = y0 + py - 4, x = x0 + px - 4;
        float v = 0.f;
        if (y >= 0 && y < H_ && x >= 0 && x < W_)
            v = f2[((size_t)(b * C_ + c) * H_ + y) * W_ + x];
        s2[(py * HALO_W + px) * C_ + c] = (_Float16)v;
    }

    // own pixel's feat1 channels, packed 2 x f16 per dword
    const int ly = tid >> 4, lx = tid & 15;
    const int y = y0 + ly, x = x0 + lx;
    unsigned f1p[32];
    for (int c2 = 0; c2 < 32; ++c2) {
        union { unsigned u; _Float16 h[2]; } pk;
        pk.h[0] = (_Float16)f1[((size_t)(b * C_ + 2 * c2 + 0) * H_ + y) * W_ + x];
        pk.h[1] = (_Float16)f1[((size_t)(b * C_ + 2 * c2 + 1) * H_ + y) * W_ + x];
        f1p[c2] = pk.u;
    }
    __syncthreads();

    _Float16* op = corr + ((size_t)(b * H_ + y) * W_ + x) * NDP;
    for (int dy = 0; dy < 9; ++dy)
        for (int dx = 0; dx < 9; ++dx) {
            const uint4* bp = (const uint4*)&s2[((ly + dy) * HALO_W + (lx + dx)) * C_];
            float acc = 0.f;
            for (int g = 0; g < 8; ++g) {
                union { uint4 u; _Float16 h[8]; } w; w.u = bp[g];
                union { unsigned u; _Float16 h[2]; } a0, a1, a2, a3;
                a0.u = f1p[g * 4 + 0]; a1.u = f1p[g * 4 + 1];
                a2.u = f1p[g * 4 + 2]; a3.u = f1p[g * 4 + 3];
                acc += (float)a0.h[0] * (float)w.h[0] + (float)a0.h[1] * (float)w.h[1]
                     + (float)a1.h[0] * (float)w.h[2] + (float)a1.h[1] * (float)w.h[3]
                     + (float)a2.h[0] * (float)w.h[4] + (float)a2.h[1] * (float)w.h[5]
                     + (float)a3.h[0] * (float)w.h[6] + (float)a3.h[1] * (float)w.h[7];
            }
            op[dy * 9 + dx] = (_Float16)(acc * 0.015625f);   // mean over 64 channels
        }
    for (int d = ND; d < NDP; ++d) op[d] = (_Float16)0.f;    // zero padded channels
}

// ---------------------------------------------------------------------------
// Weight pre-pack: fp32 OIHW -> f16 WMMA A-fragment order (16x32 per chunk).
// ISA layout (16-bit A 16x32): lanes 0-15 hold M=lane, K pairs
//   V0:K0,1 V1:K2,3 V2:K4,5 V3:K6,7 V4:K16,17 .. V7:K22,23
// lanes 16-31 hold M=lane-16, K shifted by 8 (V0:K8,9 .. V7:K30,31).
// k0(lane,v) = (v>>2)*16 + (lane>>4)*8 + (v&3)*2
// Linear index: ((((ct*3+ky)*3+kx)*KC + kc)*256 + lane*8 + v)
// ---------------------------------------------------------------------------
__global__ __launch_bounds__(256) void pack_weights(const float* __restrict__ w,
                                                    uint32_t* __restrict__ packed,
                                                    int Cout, int Cinl, int Cinp, int total)
{
    int tid = blockIdx.x * 256 + threadIdx.x;
    if (tid >= total) return;
    int v    = tid & 7;
    int lane = (tid >> 3) & 31;
    int q    = tid >> 8;
    int KC   = Cinp >> 5;
    int kc = q % KC; q /= KC;
    int kx = q % 3;  q /= 3;
    int ky = q % 3;
    int ct = q / 3;
    int m  = ct * 16 + (lane & 15);
    int k0 = ((v >> 2) << 4) + ((lane >> 4) << 3) + ((v & 3) << 1);
    int c0 = (kc << 5) + k0;
    union { uint32_t u; _Float16 h[2]; } pk;
    float e0 = (m < Cout && c0     < Cinl) ? w[((m * Cinl + c0    ) * 3 + ky) * 3 + kx] : 0.f;
    float e1 = (m < Cout && c0 + 1 < Cinl) ? w[((m * Cinl + c0 + 1) * 3 + ky) * 3 + kx] : 0.f;
    pk.h[0] = (_Float16)e0; pk.h[1] = (_Float16)e1;
    packed[tid] = pk.u;
}

// ---------------------------------------------------------------------------
// 3x3 conv as implicit GEMM on WMMA f32_16x16x32_f16.
//   D[m=Cout tile 16][n=16 pixels] += A[m][k] * B[k][n], K = Cin*9
// Activations NHWC f16 (Cin multiple of 32). Workgroup = 256 thr (8 waves),
// spatial tile 32x4 (+1 halo each side) staged in LDS via CDNA5 async
// global->LDS DMA (ASYNCcnt) when available; each wave keeps one 16-pixel
// tile (B fragments are re-read from LDS for every Cout tile) and streams the
// pre-packed A fragments from L2. Epilogue: bias+ReLU packed into a single
// b128 store per lane (8 consecutive NHWC channels).
// ---------------------------------------------------------------------------
template<int CIN, int COUT, bool FINAL>
__global__ __launch_bounds__(256) void conv3x3_wmma(const _Float16* __restrict__ in,
                                                    const uint32_t* __restrict__ wpack,
                                                    const float* __restrict__ bias,
                                                    _Float16* __restrict__ outh,
                                                    float* __restrict__ outf)
{
    const int TW = 32, TH = 4, HW = TW + 2, HH = TH + 2;
    const int KC = CIN / 32, CT = COUT / 16;
    __shared__ __align__(16) _Float16 lds[HH * HW * CIN];   // <= 52224 B (CIN=128)

    const int b  = blockIdx.z;
    const int y0 = blockIdx.y * TH;
    const int x0 = blockIdx.x * TW;
    const int tid = threadIdx.x;

    __builtin_prefetch(wpack, 0, 3);   // warm the weight stream (global_prefetch)

    // stage input tile + halo, zero-filling out-of-range (SAME padding)
#if ASYNC_STAGE
    // builtin signature (from compiler diagnostic): param0 = v4i addrspace(1)*,
    // param1 = LDS-side v4i pointer; imm offset, imm cpol.
    typedef __attribute__((address_space(1))) v4i_ gv4i;
    typedef __attribute__((address_space(3))) v4i_ lv4i;
    for (int i = tid; i < HH * HW * (CIN / 8); i += 256) {
        int cc = i % (CIN / 8);
        int p  = i / (CIN / 8);
        int px = p % HW, py = p / HW;
        int y = y0 + py - 1, x = x0 + px - 1;
        _Float16* ldst = lds + p * CIN + cc * 8;
        if (y >= 0 && y < H_ && x >= 0 && x < W_) {
            const _Float16* gsrc = in + ((size_t)(b * H_ + y) * W_ + x) * CIN + cc * 8;
            __builtin_amdgcn_global_load_async_to_lds_b128(
                (gv4i*)(uintptr_t)gsrc, (lv4i*)(uint32_t)(uintptr_t)ldst, 0, 0);
        } else {
            uint4 z; z.x = z.y = z.z = z.w = 0u;
            *(uint4*)ldst = z;
        }
    }
    __builtin_amdgcn_s_wait_asynccnt(0);
    __syncthreads();
#else
    for (int i = tid; i < HH * HW * (CIN / 8); i += 256) {
        int cc = i % (CIN / 8);
        int p  = i / (CIN / 8);
        int px = p % HW, py = p / HW;
        int y = y0 + py - 1, x = x0 + px - 1;
        uint4 v; v.x = v.y = v.z = v.w = 0u;
        if (y >= 0 && y < H_ && x >= 0 && x < W_)
            v = *(const uint4*)(in + ((size_t)(b * H_ + y) * W_ + x) * CIN + cc * 8);
        *(uint4*)(lds + p * CIN + cc * 8) = v;
    }
    __syncthreads();
#endif

    const int lane = tid & 31, wave = tid >> 5;
    const int h = lane >> 4, n = lane & 15;

    for (int t = wave; t < CT * 8; t += 8) {
        int ct  = t >> 3;
        int pt  = t & 7;
        int ly  = pt >> 1;
        int lx0 = (pt & 1) * 16;
        v8f acc = {};
        for (int ky = 0; ky < 3; ++ky)
            for (int kx = 0; kx < 3; ++kx) {
                const _Float16* bbase =
                    lds + ((ly + ky) * HW + (lx0 + n + kx)) * CIN + h * 8;
                const uint32_t* abase =
                    wpack + ((((ct * 3 + ky) * 3 + kx) * KC) << 8) + (lane << 3);
                for (int kc = 0; kc < KC; ++kc) {
                    union { uint4 u[2]; v16h v; } A, Bm;
                    const uint4* ap = (const uint4*)(abase + (kc << 8));
                    A.u[0] = ap[0]; A.u[1] = ap[1];
                    const uint4* bp = (const uint4*)(bbase + kc * 32);
                    Bm.u[0] = bp[0]; Bm.u[1] = bp[2];
                    acc = __builtin_amdgcn_wmma_f32_16x16x32_f16(
                        false, A.v, false, Bm.v, (short)0, acc, false, false);
                }
            }
        // C/D layout: VGPR r, lanes 0-15 -> M=r, N=lane; lanes 16-31 -> M=r+8
        int yy = y0 + ly, xx = x0 + lx0 + n;
        int m0 = ct * 16 + (h << 3);            // 8 consecutive Cout channels
        if (FINAL) {
            // only m < 2 valid (weights padded with zero rows), fp32 NCHW out
            if (h == 0) {
                outf[((size_t)(b * 2 + 0) * H_ + yy) * W_ + xx] = acc[0] + bias[0];
                outf[((size_t)(b * 2 + 1) * H_ + yy) * W_ + xx] = acc[1] + bias[1];
            }
        } else {
            union { uint4 u; _Float16 hh[8]; } st;
#pragma unroll
            for (int r = 0; r < 8; ++r) {
                float v = acc[r] + bias[m0 + r];
                st.hh[r] = (_Float16)fmaxf(v, 0.f);
            }
            *(uint4*)(outh + ((size_t)(b * H_ + yy) * W_ + xx) * COUT + m0) = st.u;
        }
    }
}

// ---------------------------------------------------------------------------
// Host-side orchestration (graph-capture safe: launches only, all on stream)
// ---------------------------------------------------------------------------
extern "C" void kernel_launch(void* const* d_in, const int* in_sizes, int n_in,
                              void* d_out, int out_size, void* d_ws, size_t ws_size,
                              hipStream_t stream) {
    const float* feat1 = (const float*)d_in[0];
    const float* feat2 = (const float*)d_in[1];
    const float* w1 = (const float*)d_in[2]; const float* b1 = (const float*)d_in[3];
    const float* w2 = (const float*)d_in[4]; const float* b2 = (const float*)d_in[5];
    const float* w3 = (const float*)d_in[6]; const float* b3 = (const float*)d_in[7];
    const float* w4 = (const float*)d_in[8]; const float* b4 = (const float*)d_in[9];
    float* out = (float*)d_out;

    // workspace layout (bytes, all 16B-aligned)
    char* ws = (char*)d_ws;
    const size_t SZ_CORR = (size_t)B_ * H_ * W_ * NDP * 2;   //  56,623,104
    const size_t SZ_X1   = (size_t)B_ * H_ * W_ * 128 * 2;   //  75,497,472
    const size_t SZ_X2   = (size_t)B_ * H_ * W_ * 64  * 2;   //  37,748,736
    const size_t SZ_X3   = (size_t)B_ * H_ * W_ * 32  * 2;   //  18,874,368
    _Float16* corr = (_Float16*)(ws);
    _Float16* x1   = (_Float16*)(ws + SZ_CORR);
    _Float16* x2   = (_Float16*)(ws + SZ_CORR + SZ_X1);
    _Float16* x3   = (_Float16*)(ws + SZ_CORR + SZ_X1 + SZ_X2);
    char* pw = ws + SZ_CORR + SZ_X1 + SZ_X2 + SZ_X3;
    const int T1 = (128/16)*9*(96/32) *256;   // 55296 dwords
    const int T2 = ( 64/16)*9*(128/32)*256;   // 36864
    const int T3 = ( 32/16)*9*(64/32) *256;   // 9216
    const int T4 = ( 16/16)*9*(32/32) *256;   //  2304
    uint32_t* pw1 = (uint32_t*)(pw);
    uint32_t* pw2 = (uint32_t*)(pw + (size_t)T1*4);
    uint32_t* pw3 = (uint32_t*)(pw + (size_t)(T1+T2)*4);
    uint32_t* pw4 = (uint32_t*)(pw + (size_t)(T1+T2+T3)*4);

    // 1) pack weights into WMMA A-fragment layout (tiny, once per launch)
    pack_weights<<<(T1+255)/256, 256, 0, stream>>>(w1, pw1, 128, ND, NDP, T1);
    pack_weights<<<(T2+255)/256, 256, 0, stream>>>(w2, pw2,  64, 128, 128, T2);
    pack_weights<<<(T3+255)/256, 256, 0, stream>>>(w3, pw3,  32,  64,  64, T3);
    pack_weights<<<(T4+255)/256, 256, 0, stream>>>(w4, pw4,   2,  32,  32, T4);

    // 2) cost volume -> NHWC f16 (padded 96 ch)
    corr_kernel<<<dim3(W_/16, H_/8, B_), 128, 0, stream>>>(feat1, feat2, corr);

    // 3) conv stack on WMMA
    dim3 cg(W_/32, H_/4, B_);
    conv3x3_wmma< 96, 128, false><<<cg, 256, 0, stream>>>(corr, pw1, b1, x1, nullptr);
    conv3x3_wmma<128,  64, false><<<cg, 256, 0, stream>>>(x1,   pw2, b2, x2, nullptr);
    conv3x3_wmma< 64,  32, false><<<cg, 256, 0, stream>>>(x2,   pw3, b3, x3, nullptr);
    conv3x3_wmma< 32,  16, true ><<<cg, 256, 0, stream>>>(x3,   pw4, b4, nullptr, out);
}
